// LovaszSoftmaxLoss_20280835572237
// MI455X (gfx1250) — compile-verified, matrix operand order checked by reading, not run on hardware
//
#include <hip/hip_runtime.h>
#include <hip/hip_bf16.h>
#include <stdint.h>

// ---------------------------------------------------------------------------
// Lovász-Softmax loss, exact, for B=8 C=19 H=W=512 on MI455X (gfx1250).
// Pipeline per call:
//   1) online softmax stats (max, denom) per pixel            [1 kernel]
//   2) per class c (sequential, keeps 16MB sort set in L2):
//        key-gen:  key = ~((bits(|fg - p_c|) << 1) | fg)      [1 kernel]
//        LSD radix sort, 4 x 8-bit passes:
//           hist -> row-scan -> bin-scan -> stable scatter    [4 kernels/pass]
//        tile fg sums via v_wmma_f32_16x16x32_f16 reduction   [1 kernel]
//        exclusive scan of tile sums (-> G)                   [1 kernel]
//        Lovász gradient dot-product, per-block partials      [1 kernel]
//   3) deterministic final reduction + masked mean            [1 kernel]
// ---------------------------------------------------------------------------

typedef __attribute__((ext_vector_type(16))) _Float16 v16h;
typedef __attribute__((ext_vector_type(8)))  float    v8f;

#define B_    8
#define C_    19
#define HW_   262144           // 512*512 = 2^18
#define N_    2097152          // B_*HW_
#define TPB   256
#define EPT   16
#define TILE  4096             // TPB*EPT
#define NB    512              // N_/TILE
#define RADIX 256

// ---------------- block-wide inclusive scan of one value per thread --------
__device__ __forceinline__ unsigned inclScan256(unsigned v, unsigned* s) {
  const int tid = threadIdx.x;
  s[tid] = v;
  __syncthreads();
  #pragma unroll
  for (int off = 1; off < 256; off <<= 1) {
    unsigned nv = s[tid] + ((tid >= off) ? s[tid - off] : 0u);
    __syncthreads();
    s[tid] = nv;
    __syncthreads();
  }
  return s[tid];
}

// ---------------- 1) per-pixel online softmax statistics -------------------
__global__ void k_softmax_stats(const float* __restrict__ logits,
                                float* __restrict__ maxv,
                                float* __restrict__ denom) {
  int p  = blockIdx.x * TPB + threadIdx.x;
  int b  = p >> 18;
  int hw = p & (HW_ - 1);
  size_t base = (size_t)b * C_ * HW_ + (size_t)hw;
  float m = -3.0e38f, s = 0.f;
  #pragma unroll
  for (int c = 0; c < C_; ++c) {
    float x  = logits[base + (size_t)c * HW_];
    float nm = fmaxf(m, x);
    s = s * __expf(m - nm) + __expf(x - nm);
    m = nm;
  }
  maxv[p]  = m;
  denom[p] = s;
}

// ---------------- 2a) per-class sortable key generation --------------------
// err in [0,1] -> float bits fit in 30 bits; pack fg in LSB (tie-invariant),
// complement so ascending radix sort == descending error sort.
__global__ void k_keygen(const float* __restrict__ logits,
                         const long long* __restrict__ labels,
                         const float* __restrict__ maxv,
                         const float* __restrict__ denom,
                         unsigned* __restrict__ keys, int c) {
  int p  = blockIdx.x * TPB + threadIdx.x;
  int b  = p >> 18;
  int hw = p & (HW_ - 1);
  float x  = logits[((size_t)b * C_ + (size_t)c) * HW_ + (size_t)hw];
  float pr = __expf(x - maxv[p]) / denom[p];
  unsigned fg = (labels[p] == (long long)c) ? 1u : 0u;
  float err = fabsf((float)fg - pr);
  keys[p] = ~((__float_as_uint(err) << 1) | fg);
}

// ---------------- 2b) radix pass: per-block digit histogram ----------------
__global__ void k_hist(const unsigned* __restrict__ keys,
                       unsigned* __restrict__ hist, int shift) {
  __shared__ unsigned h[RADIX];
  const int tid = threadIdx.x, blk = blockIdx.x;
  h[tid] = 0;
  __syncthreads();
  size_t base = (size_t)blk * TILE;
  #pragma unroll
  for (int it = 0; it < EPT; ++it) {
    unsigned k = keys[base + it * TPB + tid];
    __builtin_prefetch(&keys[base + (it + 1) * TPB + tid], 0, 0);
    atomicAdd(&h[(k >> shift) & 255u], 1u);
  }
  __syncthreads();
  hist[(size_t)tid * NB + blk] = h[tid];   // bin-major for cross-block scan
}

// ---------------- 2c) exclusive scan of each bin's row across blocks -------
__global__ void k_rowscan(unsigned* __restrict__ hist,
                          unsigned* __restrict__ binTotal) {
  __shared__ unsigned s[NB];
  const int tid = threadIdx.x, bin = blockIdx.x;
  unsigned o0 = hist[(size_t)bin * NB + tid];
  unsigned o1 = hist[(size_t)bin * NB + tid + TPB];
  s[tid] = o0; s[tid + TPB] = o1;
  __syncthreads();
  for (int off = 1; off < NB; off <<= 1) {
    unsigned v0 = s[tid] + ((tid >= off) ? s[tid - off] : 0u);
    unsigned v1 = s[tid + TPB] + (((tid + TPB) >= off) ? s[tid + TPB - off] : 0u);
    __syncthreads();
    s[tid] = v0; s[tid + TPB] = v1;
    __syncthreads();
  }
  hist[(size_t)bin * NB + tid]       = s[tid] - o0;
  hist[(size_t)bin * NB + tid + TPB] = s[tid + TPB] - o1;
  if (tid == TPB - 1) binTotal[bin] = s[NB - 1];
}

// ---------------- 2d) exclusive scan of the 256 bin totals -----------------
__global__ void k_binscan(const unsigned* __restrict__ binTotal,
                          unsigned* __restrict__ binBase) {
  __shared__ unsigned s[RADIX];
  unsigned v = binTotal[threadIdx.x];
  unsigned incl = inclScan256(v, s);
  binBase[threadIdx.x] = incl - v;
}

// ---------------- 2e) stable scatter using wave32 ballot ranking -----------
__global__ void k_scatter(const unsigned* __restrict__ in,
                          unsigned* __restrict__ out,
                          const unsigned* __restrict__ hist,
                          const unsigned* __restrict__ binBase, int shift) {
  __shared__ unsigned binOffset[RADIX];
  __shared__ unsigned waveCnt[8 * RADIX];
  const int tid = threadIdx.x, blk = blockIdx.x;
  const int wave = tid >> 5, lane = tid & 31;
  const unsigned laneLT = (1u << lane) - 1u;
  binOffset[tid] = binBase[tid] + hist[(size_t)tid * NB + blk];
  size_t base = (size_t)blk * TILE;
  for (int it = 0; it < EPT; ++it) {
    #pragma unroll
    for (int j = 0; j < 8; ++j) waveCnt[j * RADIX + tid] = 0;
    __syncthreads();                       // also covers binOffset init
    unsigned key = in[base + it * TPB + tid];
    __builtin_prefetch(&in[base + (it + 1) * TPB + tid], 0, 0);
    unsigned d = (key >> shift) & 255u;
    unsigned mask = 0xffffffffu;           // lanes in my wave with same digit
    #pragma unroll
    for (int b = 0; b < 8; ++b) {
      unsigned bal = __builtin_amdgcn_ballot_w32(((d >> b) & 1u) != 0u);
      mask &= ((d >> b) & 1u) ? bal : ~bal;
    }
    unsigned rank = __popc(mask & laneLT);
    if (rank == 0) waveCnt[wave * RADIX + d] = __popc(mask);
    __syncthreads();
    unsigned pre = 0;                      // same-digit count in earlier waves
    for (int w = 0; w < wave; ++w) pre += waveCnt[w * RADIX + d];
    out[binOffset[d] + pre + rank] = key;  // stable: (blk, iter, wave, lane)
    __syncthreads();
    unsigned add = 0;
    #pragma unroll
    for (int w = 0; w < 8; ++w) add += waveCnt[w * RADIX + tid];
    binOffset[tid] += add;
    __syncthreads();
  }
}

// ---------------- 2f) per-tile fg sums via WMMA matrix-pipe reduction ------
// B operand = 512 fg values (0/1) of one wave; A = all-ones. Every D row is
// the column-sum vector, and d[0] across the 32 lanes covers each column
// exactly twice (M=0 and M=8 per the C/D layout), so sum(d[0]) = 2 * total.
// Exact: column sums <= 32 in f16/f32.
__global__ void k_tilefg(const unsigned* __restrict__ keys,
                         unsigned* __restrict__ tileFg) {
  __shared__ float red[TPB];
  const int tid = threadIdx.x, blk = blockIdx.x;
  size_t base = (size_t)blk * TILE;
  v16h bfrag;
  #pragma unroll
  for (int it = 0; it < EPT; ++it) {
    unsigned k = keys[base + it * TPB + tid];
    bfrag[it] = (_Float16)(float)((~k) & 1u);
  }
  v16h aones;
  #pragma unroll
  for (int j = 0; j < 16; ++j) aones[j] = (_Float16)1.0f;
  v8f czero = {};
  v8f d = __builtin_amdgcn_wmma_f32_16x16x32_f16(
      false, aones, false, bfrag, (short)0, czero, false, false);
  red[tid] = d[0];
  __syncthreads();
  for (int s = TPB >> 1; s > 0; s >>= 1) {
    if (tid < s) red[tid] += red[tid + s];
    __syncthreads();
  }
  if (tid == 0) tileFg[blk] = (unsigned)(red[0] * 0.5f + 0.25f);
}

// ---------------- 2g) exclusive scan of tile fg sums; total = G ------------
__global__ void k_tilefgscan(const unsigned* __restrict__ tileFg,
                             unsigned* __restrict__ tileFgBase,
                             unsigned* __restrict__ Gout) {
  __shared__ unsigned s[NB];
  const int tid = threadIdx.x;
  unsigned o0 = tileFg[tid], o1 = tileFg[tid + TPB];
  s[tid] = o0; s[tid + TPB] = o1;
  __syncthreads();
  for (int off = 1; off < NB; off <<= 1) {
    unsigned v0 = s[tid] + ((tid >= off) ? s[tid - off] : 0u);
    unsigned v1 = s[tid + TPB] + s[tid + TPB - off];
    __syncthreads();
    s[tid] = v0; s[tid + TPB] = v1;
    __syncthreads();
  }
  tileFgBase[tid]       = s[tid] - o0;
  tileFgBase[tid + TPB] = s[tid + TPB] - o1;
  if (tid == TPB - 1) *Gout = s[NB - 1];
}

// ---------------- 2h) Lovász gradient dot product, per-block partials ------
__global__ void k_loss(const unsigned* __restrict__ keys,
                       const unsigned* __restrict__ tileFgBase,
                       const unsigned* __restrict__ Gp,
                       float* __restrict__ lossBlock) {
  __shared__ unsigned su[TPB];
  __shared__ float rf[TPB];
  const int tid = threadIdx.x, blk = blockIdx.x;
  const unsigned G = *Gp;
  size_t base = (size_t)blk * TILE + (size_t)tid * EPT;
  unsigned k[EPT];
  const uint4* kp = (const uint4*)(keys + base);
  #pragma unroll
  for (int q = 0; q < 4; ++q) {
    uint4 v = kp[q];
    k[q * 4 + 0] = v.x; k[q * 4 + 1] = v.y;
    k[q * 4 + 2] = v.z; k[q * 4 + 3] = v.w;
  }
  unsigned pf[EPT]; unsigned run = 0;
  #pragma unroll
  for (int j = 0; j < EPT; ++j) { run += (~k[j]) & 1u; pf[j] = run; }
  unsigned incl = inclScan256(run, su);
  unsigned cbase = tileFgBase[blk] + (incl - run);
  float contrib = 0.f;
  if (G > 0u) {                            // absent class contributes 0
    const float fG = (float)G;
    #pragma unroll
    for (int j = 0; j < EPT; ++j) {
      unsigned ok = ~k[j];
      float fg = (float)(ok & 1u);
      float e  = __uint_as_float(ok >> 1);
      float n  = (float)(base + (size_t)j + 1);   // 1-indexed rank (desc err)
      float cf = (float)(cbase + pf[j]);          // inclusive cumFG
      float Jn = 1.f - (fG - cf) / (fG + n - cf);
      float cp = cf - fg, np = n - 1.f;
      float Jp = 1.f - (fG - cp) / (fG + np - cp);  // J(0)=0 naturally
      contrib += e * (Jn - Jp);
    }
  }
  rf[tid] = contrib;
  __syncthreads();
  for (int s2 = TPB >> 1; s2 > 0; s2 >>= 1) {
    if (tid < s2) rf[tid] += rf[tid + s2];
    __syncthreads();
  }
  if (tid == 0) lossBlock[blk] = rf[0];
}

// ---------------- 3) deterministic final reduction + masked mean -----------
__global__ void k_final(const float* __restrict__ lossBlock,
                        const unsigned* __restrict__ G,
                        float* __restrict__ out) {
  __shared__ float rf[TPB];
  const int tid = threadIdx.x;
  float accL = 0.f, accP = 0.f;
  for (int c = 0; c < C_; ++c) {
    float s = 0.f;
    for (int j = tid; j < NB; j += TPB) s += lossBlock[(size_t)c * NB + j];
    rf[tid] = s;
    __syncthreads();
    for (int s2 = TPB >> 1; s2 > 0; s2 >>= 1) {
      if (tid < s2) rf[tid] += rf[tid + s2];
      __syncthreads();
    }
    if (tid == 0 && G[c] > 0u) { accL += rf[0]; accP += 1.f; }
    __syncthreads();
  }
  if (tid == 0) out[0] = accL / fmaxf(accP, 1.f);
}

// ---------------------------------------------------------------------------
extern "C" void kernel_launch(void* const* d_in, const int* in_sizes, int n_in,
                              void* d_out, int out_size, void* d_ws, size_t ws_size,
                              hipStream_t stream) {
  (void)in_sizes; (void)n_in; (void)out_size; (void)ws_size;
  const float*     logits = (const float*)d_in[0];
  const long long* labels = (const long long*)d_in[1];
  float* out = (float*)d_out;

  char* w = (char*)d_ws;
  size_t off = 0;
  #define CARVE(ty, name, bytes) \
    ty name = (ty)(w + off); off += (((size_t)(bytes)) + 255u) & ~(size_t)255u;
  CARVE(float*,    maxv,       (size_t)N_ * 4)            // 8 MB
  CARVE(float*,    denom,      (size_t)N_ * 4)            // 8 MB
  CARVE(unsigned*, keysA,      (size_t)N_ * 4)            // 8 MB
  CARVE(unsigned*, keysB,      (size_t)N_ * 4)            // 8 MB
  CARVE(unsigned*, hist,       (size_t)RADIX * NB * 4)    // 512 KB
  CARVE(unsigned*, binTotal,   (size_t)RADIX * 4)
  CARVE(unsigned*, binBase,    (size_t)RADIX * 4)
  CARVE(unsigned*, tileFg,     (size_t)NB * 4)
  CARVE(unsigned*, tileFgBase, (size_t)NB * 4)
  CARVE(unsigned*, perG,       (size_t)C_ * 4)
  CARVE(float*,    lossBlock,  (size_t)C_ * NB * 4)       // 38 KB
  #undef CARVE

  k_softmax_stats<<<N_ / TPB, TPB, 0, stream>>>(logits, maxv, denom);

  for (int c = 0; c < C_; ++c) {
    k_keygen<<<N_ / TPB, TPB, 0, stream>>>(logits, labels, maxv, denom, keysA, c);
    unsigned* pin = keysA; unsigned* pout = keysB;
    for (int pass = 0; pass < 4; ++pass) {
      const int shift = pass * 8;
      k_hist   <<<NB,    TPB, 0, stream>>>(pin, hist, shift);
      k_rowscan<<<RADIX, TPB, 0, stream>>>(hist, binTotal);
      k_binscan<<<1,     TPB, 0, stream>>>(binTotal, binBase);
      k_scatter<<<NB,    TPB, 0, stream>>>(pin, pout, hist, binBase, shift);
      unsigned* t = pin; pin = pout; pout = t;
    }
    // after 4 passes, sorted keys are back in keysA (== pin)
    k_tilefg    <<<NB, TPB, 0, stream>>>(pin, tileFg);
    k_tilefgscan<<<1,  TPB, 0, stream>>>(tileFg, tileFgBase, perG + c);
    k_loss      <<<NB, TPB, 0, stream>>>(pin, tileFgBase, perG + c,
                                         lossBlock + (size_t)c * NB);
  }
  k_final<<<1, TPB, 0, stream>>>(lossBlock, perG, out);
}
